// NeuronLlamaMLP_45956150067864
// MI455X (gfx1250) — compile-verified, matrix operand order checked by reading, not run on hardware
//
#include <hip/hip_runtime.h>
#include <cstdint>

#define RMS_EPS 1e-5f

typedef __attribute__((ext_vector_type(16))) __bf16 v16bf;
typedef __attribute__((ext_vector_type(8)))  float  v8f;

union FragBF16 {
    uint4 q[2];   // 32 bytes
    v16bf v;      // 16 x bf16
};

// f32 pair -> packed bf16 (round-half-up) in 3 VALU ops:
// two v_add_u32 (rounding bias) + one v_perm_b32 (select high bytes).
__device__ __forceinline__ unsigned pack_bf16(float a, float b) {
    unsigned ua = __float_as_uint(a) + 0x8000u;
    unsigned ub = __float_as_uint(b) + 0x8000u;
    return __builtin_amdgcn_perm(ub, ua, 0x07060302u);
}

__device__ __forceinline__ unsigned short bf16_one(float a) {
    return (unsigned short)((__float_as_uint(a) + 0x8000u) >> 16);
}

// convert 16 consecutive floats (4 x float4 in regs) -> 2 x uint4 of bf16
__device__ __forceinline__ void cvt16(const float4& a, const float4& b,
                                      const float4& c, const float4& d,
                                      uint4& lo, uint4& hi) {
    lo.x = pack_bf16(a.x, a.y); lo.y = pack_bf16(a.z, a.w);
    lo.z = pack_bf16(b.x, b.y); lo.w = pack_bf16(b.z, b.w);
    hi.x = pack_bf16(c.x, c.y); hi.y = pack_bf16(c.z, c.w);
    hi.z = pack_bf16(d.x, d.y); hi.w = pack_bf16(d.z, d.w);
}

// ---------------------------------------------------------------------------
// Kernel 1: RMSNorm, fp32 [S,H] -> bf16 [S,H].  One block (256 thr) per row.
// ---------------------------------------------------------------------------
__global__ void __launch_bounds__(256)
rmsnorm_bf16_kernel(const float* __restrict__ x, const float* __restrict__ w,
                    unsigned short* __restrict__ h, int H) {
    const int s = blockIdx.x;
    const int t = threadIdx.x;
    const float4* xr = (const float4*)(x + (size_t)s * H);
    const int chunks = H >> 10;  // H / (256 threads * 4 floats)

    float ss = 0.f;
    for (int c = 0; c < chunks; ++c) {
        float4 v = xr[c * 256 + t];
        ss += v.x * v.x + v.y * v.y + v.z * v.z + v.w * v.w;
    }
    __shared__ float red[256];
    red[t] = ss;
    __syncthreads();
    for (int off = 128; off > 0; off >>= 1) {
        if (t < off) red[t] += red[t + off];
        __syncthreads();
    }
    const float scale = __frsqrt_rn(red[0] / (float)H + RMS_EPS);

    const float4* wr = (const float4*)w;
    uint2* ho = (uint2*)(h + (size_t)s * H);
    for (int c = 0; c < chunks; ++c) {
        int idx = c * 256 + t;
        float4 xv = xr[idx];
        float4 wv = wr[idx];
        uint2 o;
        o.x = pack_bf16(xv.x * scale * wv.x, xv.y * scale * wv.y);
        o.y = pack_bf16(xv.z * scale * wv.z, xv.w * scale * wv.w);
        ho[idx] = o;
    }
}

// ---------------------------------------------------------------------------
// Kernel 2: fused gate/up GEMM + SwiGLU, double-buffered LDS.
//   g = h @ gate_w^T ; u = h @ up_w^T ; act = silu(g)*u   (bf16 out)
//   Grid: x = M-blocks (fast, few -> weight strips stay hot in L2), y = N.
//   Block tile 128(M) x 128(N), K-tile 32.  8 waves; wave tile 32x64.
// ---------------------------------------------------------------------------
#define TSZ (128 * 32)

__global__ void __launch_bounds__(256)
gateup_swiglu_kernel(const unsigned short* __restrict__ hbf,
                     const float* __restrict__ gate_w,
                     const float* __restrict__ up_w,
                     unsigned short* __restrict__ act,
                     int H, int I) {
    __shared__ __align__(16) unsigned short sA[2 * TSZ];
    __shared__ __align__(16) unsigned short sG[2 * TSZ];
    __shared__ __align__(16) unsigned short sU[2 * TSZ];

    const int t    = threadIdx.x;
    const int lane = t & 31;
    const int wv   = t >> 5;
    const int wm   = (wv & 3) * 32;   // wave M offset inside block tile
    const int wn   = (wv >> 2) * 64;  // wave N offset inside block tile
    const int mBlock = blockIdx.x * 128;   // M fast-moving
    const int nBlock = blockIdx.y * 128;

    const v8f vzero = {0.f, 0.f, 0.f, 0.f, 0.f, 0.f, 0.f, 0.f};
    v8f accG[2][4], accU[2][4];
#pragma unroll
    for (int mi = 0; mi < 2; ++mi)
#pragma unroll
        for (int ni = 0; ni < 4; ++ni) { accG[mi][ni] = vzero; accU[mi][ni] = vzero; }

    const int row = t >> 1;          // 128 rows, 2 threads per row
    const int seg = (t & 1) * 16;    // each thread: 16 K-elements
    const int ldsOff = row * 32 + seg;

    const int frow = lane & 15;          // fragment row/col within 16-tile
    const int kh   = (lane >> 4) * 8;    // K-half select per ISA layout

    const size_t aRow = (size_t)(mBlock + row) * H;
    const size_t wRow = (size_t)(nBlock + row) * H;

    // ---- preload K-tile 0 into buffer 0 ----
    {
        const uint4* as = (const uint4*)(hbf + aRow + seg);
        ((uint4*)&sA[ldsOff])[0] = as[0];
        ((uint4*)&sA[ldsOff])[1] = as[1];
        const float4* gs = (const float4*)(gate_w + wRow + seg);
        const float4* us = (const float4*)(up_w   + wRow + seg);
        uint4 lo, hi;
        cvt16(gs[0], gs[1], gs[2], gs[3], lo, hi);
        ((uint4*)&sG[ldsOff])[0] = lo; ((uint4*)&sG[ldsOff])[1] = hi;
        cvt16(us[0], us[1], us[2], us[3], lo, hi);
        ((uint4*)&sU[ldsOff])[0] = lo; ((uint4*)&sU[ldsOff])[1] = hi;
    }
    __syncthreads();

    const int kIters = H >> 5;
    for (int kt = 0; kt < kIters; ++kt) {
        const unsigned short* bufA = sA + (kt & 1) * TSZ;
        const unsigned short* bufG = sG + (kt & 1) * TSZ;
        const unsigned short* bufU = sU + (kt & 1) * TSZ;
        const bool hasNext = (kt + 1) < kIters;

        // ---- issue next tile's global loads (results consumed after WMMAs) ----
        uint4  nA0 = {}, nA1 = {};
        float4 g0 = {}, g1 = {}, g2 = {}, g3 = {};
        float4 u0 = {}, u1 = {}, u2 = {}, u3 = {};
        if (hasNext) {
            const int kn = (kt + 1) * 32;
            const uint4* as = (const uint4*)(hbf + aRow + kn + seg);
            nA0 = as[0]; nA1 = as[1];
            const float4* gs = (const float4*)(gate_w + wRow + kn + seg);
            const float4* us = (const float4*)(up_w   + wRow + kn + seg);
            __builtin_prefetch((const void*)(gs + 8), 0, 0);  // K-tile after next
            __builtin_prefetch((const void*)(us + 8), 0, 0);
            g0 = gs[0]; g1 = gs[1]; g2 = gs[2]; g3 = gs[3];
            u0 = us[0]; u1 = us[1]; u2 = us[2]; u3 = us[3];
        }

        // ---- fragments from current buffer ----
        FragBF16 aF[2], gF[4], uF[4];
#pragma unroll
        for (int mi = 0; mi < 2; ++mi) {
            const unsigned short* b = &bufA[(wm + mi * 16 + frow) * 32 + kh];
            aF[mi].q[0] = *(const uint4*)b;
            aF[mi].q[1] = *(const uint4*)(b + 16);
        }
#pragma unroll
        for (int ni = 0; ni < 4; ++ni) {
            const unsigned short* bg = &bufG[(wn + ni * 16 + frow) * 32 + kh];
            gF[ni].q[0] = *(const uint4*)bg;
            gF[ni].q[1] = *(const uint4*)(bg + 16);
            const unsigned short* bu = &bufU[(wn + ni * 16 + frow) * 32 + kh];
            uF[ni].q[0] = *(const uint4*)bu;
            uF[ni].q[1] = *(const uint4*)(bu + 16);
        }

        // ---- 16 WMMAs ----
#pragma unroll
        for (int mi = 0; mi < 2; ++mi)
#pragma unroll
            for (int ni = 0; ni < 4; ++ni) {
                accG[mi][ni] = __builtin_amdgcn_wmma_f32_16x16x32_bf16(
                    false, aF[mi].v, false, gF[ni].v, (short)0, accG[mi][ni], false, false);
                accU[mi][ni] = __builtin_amdgcn_wmma_f32_16x16x32_bf16(
                    false, aF[mi].v, false, uF[ni].v, (short)0, accU[mi][ni], false, false);
            }

        // ---- convert + store next tile into the other buffer ----
        if (hasNext) {
            unsigned short* nA = sA + ((kt + 1) & 1) * TSZ;
            unsigned short* nG = sG + ((kt + 1) & 1) * TSZ;
            unsigned short* nU = sU + ((kt + 1) & 1) * TSZ;
            ((uint4*)&nA[ldsOff])[0] = nA0;
            ((uint4*)&nA[ldsOff])[1] = nA1;
            uint4 lo, hi;
            cvt16(g0, g1, g2, g3, lo, hi);
            ((uint4*)&nG[ldsOff])[0] = lo; ((uint4*)&nG[ldsOff])[1] = hi;
            cvt16(u0, u1, u2, u3, lo, hi);
            ((uint4*)&nU[ldsOff])[0] = lo; ((uint4*)&nU[ldsOff])[1] = hi;
        }
        __syncthreads();
    }

    // ---- epilogue: silu(g)*u -> bf16 ----
    const int mLane = (lane >> 4) * 8;
    const int nLane = lane & 15;
#pragma unroll
    for (int mi = 0; mi < 2; ++mi)
#pragma unroll
        for (int ni = 0; ni < 4; ++ni) {
            const int n = nBlock + wn + ni * 16 + nLane;
#pragma unroll
            for (int r = 0; r < 8; ++r) {
                float g = accG[mi][ni][r];
                float u = accU[mi][ni][r];
                float a = (g / (1.f + __expf(-g))) * u;
                const int m = mBlock + wm + mi * 16 + mLane + r;
                act[(size_t)m * I + n] = bf16_one(a);
            }
        }
}

// ---------------------------------------------------------------------------
// Kernel 3: down projection, double-buffered.  out = act @ down_w^T (fp32 out)
// ---------------------------------------------------------------------------
__global__ void __launch_bounds__(256)
down_proj_kernel(const unsigned short* __restrict__ act,
                 const float* __restrict__ down_w,
                 float* __restrict__ out,
                 int I, int H) {
    __shared__ __align__(16) unsigned short sA[2 * TSZ];
    __shared__ __align__(16) unsigned short sB[2 * TSZ];

    const int t    = threadIdx.x;
    const int lane = t & 31;
    const int wv   = t >> 5;
    const int wm   = (wv & 3) * 32;
    const int wn   = (wv >> 2) * 64;
    const int mBlock = blockIdx.x * 128;   // M fast-moving
    const int nBlock = blockIdx.y * 128;

    const v8f vzero = {0.f, 0.f, 0.f, 0.f, 0.f, 0.f, 0.f, 0.f};
    v8f acc[2][4];
#pragma unroll
    for (int mi = 0; mi < 2; ++mi)
#pragma unroll
        for (int ni = 0; ni < 4; ++ni) acc[mi][ni] = vzero;

    const int row = t >> 1;
    const int seg = (t & 1) * 16;
    const int ldsOff = row * 32 + seg;
    const int frow = lane & 15;
    const int kh   = (lane >> 4) * 8;

    const size_t aRow = (size_t)(mBlock + row) * I;
    const size_t wRow = (size_t)(nBlock + row) * I;

    {
        const uint4* as = (const uint4*)(act + aRow + seg);
        ((uint4*)&sA[ldsOff])[0] = as[0];
        ((uint4*)&sA[ldsOff])[1] = as[1];
        const float4* ws = (const float4*)(down_w + wRow + seg);
        uint4 lo, hi;
        cvt16(ws[0], ws[1], ws[2], ws[3], lo, hi);
        ((uint4*)&sB[ldsOff])[0] = lo; ((uint4*)&sB[ldsOff])[1] = hi;
    }
    __syncthreads();

    const int kIters = I >> 5;
    for (int kt = 0; kt < kIters; ++kt) {
        const unsigned short* bufA = sA + (kt & 1) * TSZ;
        const unsigned short* bufB = sB + (kt & 1) * TSZ;
        const bool hasNext = (kt + 1) < kIters;

        uint4  nA0 = {}, nA1 = {};
        float4 w0 = {}, w1 = {}, w2 = {}, w3 = {};
        if (hasNext) {
            const int kn = (kt + 1) * 32;
            const uint4* as = (const uint4*)(act + aRow + kn + seg);
            nA0 = as[0]; nA1 = as[1];
            const float4* ws = (const float4*)(down_w + wRow + kn + seg);
            __builtin_prefetch((const void*)(ws + 8), 0, 0);
            w0 = ws[0]; w1 = ws[1]; w2 = ws[2]; w3 = ws[3];
        }

        FragBF16 aF[2], bF[4];
#pragma unroll
        for (int mi = 0; mi < 2; ++mi) {
            const unsigned short* b = &bufA[(wm + mi * 16 + frow) * 32 + kh];
            aF[mi].q[0] = *(const uint4*)b;
            aF[mi].q[1] = *(const uint4*)(b + 16);
        }
#pragma unroll
        for (int ni = 0; ni < 4; ++ni) {
            const unsigned short* b = &bufB[(wn + ni * 16 + frow) * 32 + kh];
            bF[ni].q[0] = *(const uint4*)b;
            bF[ni].q[1] = *(const uint4*)(b + 16);
        }
#pragma unroll
        for (int mi = 0; mi < 2; ++mi)
#pragma unroll
            for (int ni = 0; ni < 4; ++ni)
                acc[mi][ni] = __builtin_amdgcn_wmma_f32_16x16x32_bf16(
                    false, aF[mi].v, false, bF[ni].v, (short)0, acc[mi][ni], false, false);

        if (hasNext) {
            unsigned short* nA = sA + ((kt + 1) & 1) * TSZ;
            unsigned short* nB = sB + ((kt + 1) & 1) * TSZ;
            ((uint4*)&nA[ldsOff])[0] = nA0;
            ((uint4*)&nA[ldsOff])[1] = nA1;
            uint4 lo, hi;
            cvt16(w0, w1, w2, w3, lo, hi);
            ((uint4*)&nB[ldsOff])[0] = lo; ((uint4*)&nB[ldsOff])[1] = hi;
        }
        __syncthreads();
    }

    const int mLane = (lane >> 4) * 8;
    const int nLane = lane & 15;
#pragma unroll
    for (int mi = 0; mi < 2; ++mi)
#pragma unroll
        for (int ni = 0; ni < 4; ++ni) {
            const int n = nBlock + wn + ni * 16 + nLane;
#pragma unroll
            for (int r = 0; r < 8; ++r) {
                const int m = mBlock + wm + mi * 16 + mLane + r;
                out[(size_t)m * H + n] = acc[mi][ni][r];
            }
        }
}

// ---------------------------------------------------------------------------
extern "C" void kernel_launch(void* const* d_in, const int* in_sizes, int n_in,
                              void* d_out, int out_size, void* d_ws, size_t ws_size,
                              hipStream_t stream) {
    const float* x  = (const float*)d_in[0];
    const float* rw = (const float*)d_in[1];
    const float* gw = (const float*)d_in[2];
    const float* uw = (const float*)d_in[3];
    const float* dw = (const float*)d_in[4];
    float* out = (float*)d_out;

    const int H = in_sizes[1];            // 4096
    const int S = in_sizes[0] / H;        // B*S = 2048
    const int I = in_sizes[2] / H;        // 14336

    unsigned short* hbf = (unsigned short*)d_ws;                       // [S,H] bf16
    unsigned short* act = (unsigned short*)((char*)d_ws +
                             (size_t)S * H * sizeof(unsigned short));  // [S,I] bf16

    rmsnorm_bf16_kernel<<<S, 256, 0, stream>>>(x, rw, hbf, H);
    // M (few blocks) fast-moving so concurrent blocks share weight strips in L2
    gateup_swiglu_kernel<<<dim3(S / 128, I / 128), 256, 0, stream>>>(hbf, gw, uw, act, H, I);
    down_proj_kernel<<<dim3(S / 128, H / 128), 256, 0, stream>>>(act, dw, out, I, H);
}
